// FP8DynamicLinear_10574209483055
// MI455X (gfx1250) — 8-bit matrix operand order repaired and checked by reading, compiled, not run on hardware
//
#include <hip/hip_runtime.h>

typedef __attribute__((ext_vector_type(16))) int   v16i;
typedef __attribute__((ext_vector_type(8)))  float v8f;
typedef __attribute__((ext_vector_type(4)))  int   v4i;
typedef __attribute__((ext_vector_type(2)))  int   v2i;
typedef __attribute__((ext_vector_type(4)))  float v4f;

#define F8_MAX 448.0f

constexpr int KDIM  = 4096;
constexpr int NOUT  = 16384;
constexpr int NROWS = 8192;

#define ASYNC_LOAD16(ldsaddr, gaddr)                                          \
  asm volatile("global_load_async_to_lds_b128 %0, %1, off"                    \
               :: "v"(ldsaddr), "v"(gaddr) : "memory")
#define WAIT_ASYNCCNT(n)                                                      \
  asm volatile("s_wait_asynccnt %0" :: "n"(n) : "memory")

__device__ inline unsigned f32_to_e4m3(float x) {
  unsigned u = __float_as_uint(x);
  unsigned s = (u >> 24) & 0x80u;
  float ax = fminf(fabsf(x), 448.0f);
  unsigned bits = __float_as_uint(ax);
  int e = (int)(bits >> 23) - 127;
  unsigned out;
  if (ax < 0.015625f) {
    out = (unsigned)__builtin_rintf(ax * 512.0f);
  } else {
    unsigned mant = bits & 0x7FFFFFu;
    mant += 0x7FFFFu + ((mant >> 20) & 1u);
    if (mant >> 23) { mant = 0; e += 1; }
    out = (unsigned)(((e + 7) << 3) | (mant >> 20));
    if (out > 0x7Eu) out = 0x7Eu;
  }
  return s | out;
}

__device__ inline unsigned pack4_fp8(float a, float b, float c, float d) {
#if __has_builtin(__builtin_amdgcn_cvt_pk_fp8_f32)
  int v = __builtin_amdgcn_cvt_pk_fp8_f32(a, b, 0, false);
  v     = __builtin_amdgcn_cvt_pk_fp8_f32(c, d, v, true);
  return (unsigned)v;
#else
  return f32_to_e4m3(a) | (f32_to_e4m3(b) << 8) |
         (f32_to_e4m3(c) << 16) | (f32_to_e4m3(d) << 24);
#endif
}

__global__ void fp8lin_init_kernel(unsigned* amax) {
  if (threadIdx.x < 2) amax[threadIdx.x] = 0u;
}

__global__ void fp8lin_amax_kernel(const float* __restrict__ x, long n,
                                   unsigned* __restrict__ amax_bits) {
  long i      = (long)blockIdx.x * blockDim.x + threadIdx.x;
  long stride = (long)gridDim.x * blockDim.x;
  float m = 0.0f;
  for (long j = i * 4; j < n; j += stride * 4) {
    v4f v = *(const v4f*)(x + j);
    m = fmaxf(m, fmaxf(fmaxf(fabsf(v.x), fabsf(v.y)),
                       fmaxf(fabsf(v.z), fabsf(v.w))));
  }
  for (int off = 16; off; off >>= 1)
    m = fmaxf(m, __shfl_xor(m, off, 32));
  __shared__ float smax[8];
  int lane = threadIdx.x & 31, w = threadIdx.x >> 5;
  if (lane == 0) smax[w] = m;
  __syncthreads();
  if (threadIdx.x == 0) {
    float b = smax[0];
    for (int k = 1; k < (int)(blockDim.x >> 5); ++k) b = fmaxf(b, smax[k]);
    atomicMax(amax_bits, __float_as_uint(b));
  }
}

__global__ void fp8lin_quant_x_kernel(const float* __restrict__ x,
                                      unsigned* __restrict__ xq,
                                      const unsigned* __restrict__ amax_bits,
                                      long n4) {
  float inv = F8_MAX / __uint_as_float(amax_bits[0]);
  long i      = (long)blockIdx.x * blockDim.x + threadIdx.x;
  long stride = (long)gridDim.x * blockDim.x;
  for (long j = i; j < n4; j += stride) {
    v4f v = *(const v4f*)(x + j * 4);
    xq[j] = pack4_fp8(v.x * inv, v.y * inv, v.z * inv, v.w * inv);
  }
}

__global__ void fp8lin_quant_w_kernel(const float* __restrict__ w,
                                      unsigned char* __restrict__ wqt,
                                      const unsigned* __restrict__ amax_bits) {
  float inv = F8_MAX / __uint_as_float(amax_bits[1]);
  __shared__ unsigned char tile[64][68];
  int n0 = blockIdx.x * 64;
  int k0 = blockIdx.y * 64;
  int tx = threadIdx.x & 15;
  int ty = threadIdx.x >> 4;
  for (int kk = 0; kk < 4; ++kk) {
    int k = ty + kk * 16;
    v4f v = *(const v4f*)(w + (long)(k0 + k) * NOUT + n0 + tx * 4);
    unsigned p = pack4_fp8(v.x * inv, v.y * inv, v.z * inv, v.w * inv);
    *(unsigned*)&tile[k][tx * 4] = p;
  }
  __syncthreads();
  int n = threadIdx.x >> 2;
  int c = threadIdx.x & 3;
  v4i o;
  for (int q = 0; q < 4; ++q) {
    unsigned b0 = tile[c * 16 + q * 4 + 0][n];
    unsigned b1 = tile[c * 16 + q * 4 + 1][n];
    unsigned b2 = tile[c * 16 + q * 4 + 2][n];
    unsigned b3 = tile[c * 16 + q * 4 + 3][n];
    o[q] = (int)(b0 | (b1 << 8) | (b2 << 16) | (b3 << 24));
  }
  *(v4i*)(wqt + (long)(n0 + n) * KDIM + k0 + c * 16) = o;
}

constexpr int BM = 128, BN = 128, BK = 128;
constexpr int LDT = BK + 16;
constexpr int NT  = KDIM / BK;

__global__ __launch_bounds__(256)
void fp8lin_gemm_kernel(const unsigned char* __restrict__ xq,
                        const unsigned char* __restrict__ wqt,
                        const float* __restrict__ bias,
                        const unsigned* __restrict__ amax_bits,
                        float* __restrict__ out) {
  __shared__ unsigned char lA[2][BM * LDT];
  __shared__ unsigned char lB[2][BN * LDT];

  const int tid  = threadIdx.x;
  const int lane = tid & 31;
  const int wave = tid >> 5;
  const int wm   = wave >> 2;
  const int wn   = wave & 3;
  const int lidx = lane & 15;
  const int lhi  = lane >> 4;

  const long tileM = (long)blockIdx.y * BM;
  const long tileN = (long)blockIdx.x * BN;

  const int sRow = tid >> 3;
  const int sCol = (tid & 7) * 16;

  const unsigned ldsA = (unsigned)(uintptr_t)&lA[0][0] + sRow * LDT + sCol;
  const unsigned ldsB = (unsigned)(uintptr_t)&lB[0][0] + sRow * LDT + sCol;
  constexpr unsigned BUFSZ = (unsigned)(BM * LDT);

  v8f acc[4][2];
  for (int i = 0; i < 4; ++i)
    for (int j = 0; j < 2; ++j)
      acc[i][j] = v8f{0.f, 0.f, 0.f, 0.f, 0.f, 0.f, 0.f, 0.f};

  auto issue_tile = [&](int kt, int buf) {
    const unsigned char* gA = xq  + (tileM + sRow) * (long)KDIM + kt * BK + sCol;
    const unsigned char* gB = wqt + (tileN + sRow) * (long)KDIM + kt * BK + sCol;
    unsigned dA = ldsA + (unsigned)buf * BUFSZ;
    unsigned dB = ldsB + (unsigned)buf * BUFSZ;
    for (int j = 0; j < 4; ++j) {
      ASYNC_LOAD16(dA, gA);
      ASYNC_LOAD16(dB, gB);
      gA += 32 * (long)KDIM;
      gB += 32 * (long)KDIM;
      dA += 32u * LDT;
      dB += 32u * LDT;
    }
  };

  issue_tile(0, 0);

  for (int kt = 0; kt < NT; ++kt) {
    if (kt + 1 < NT) {
      issue_tile(kt + 1, (kt + 1) & 1);
      WAIT_ASYNCCNT(8);
    } else {
      WAIT_ASYNCCNT(0);
    }
    __syncthreads();

    const unsigned char* bufA = lA[kt & 1];
    const unsigned char* bufB = lB[kt & 1];

    v16i bfrag[2];
    for (int nf = 0; nf < 2; ++nf) {
      const unsigned char* p = bufB + (wn * 32 + nf * 16 + lidx) * LDT + lhi * 16;
      for (int q = 0; q < 4; ++q) {
        v4i t = *(const v4i*)(p + q * 32);
        bfrag[nf][4 * q + 0] = t.x;
        bfrag[nf][4 * q + 1] = t.y;
        bfrag[nf][4 * q + 2] = t.z;
        bfrag[nf][4 * q + 3] = t.w;
      }
    }

    for (int mf = 0; mf < 4; ++mf) {
      v16i afrag;
      const unsigned char* p = bufA + (wm * 64 + mf * 16 + lidx) * LDT + lhi * 16;
      for (int b = 0; b < 2; ++b)
        for (int g = 0; g < 4; ++g) {
          v2i t = *(const v2i*)(p + b * 64 + (g >> 1) * 32 + (g & 1) * 8);
          afrag[8 * b + 2 * g + 0] = t.x;
          afrag[8 * b + 2 * g + 1] = t.y;
        }
      for (int nf = 0; nf < 2; ++nf)
        acc[mf][nf] = __builtin_amdgcn_wmma_f32_16x16x128_fp8_fp8(
            afrag, bfrag[nf], (short)0, acc[mf][nf], false, false);
    }

    __syncthreads();
  }

  float scale = (__uint_as_float(amax_bits[0]) * (1.0f / F8_MAX)) *
                (__uint_as_float(amax_bits[1]) * (1.0f / F8_MAX));
  for (int nf = 0; nf < 2; ++nf) {
    long n = tileN + wn * 32 + nf * 16 + lidx;
    float bv = bias[n];
    for (int mf = 0; mf < 4; ++mf) {
      long mBase = tileM + wm * 64 + mf * 16 + lhi * 8;
      for (int r = 0; r < 8; ++r)
        out[(mBase + r) * (long)NOUT + n] = acc[mf][nf][r] * scale + bv;
    }
  }
}

extern "C" void kernel_launch(void* const* d_in, const int* in_sizes, int n_in,
                              void* d_out, int out_size, void* d_ws, size_t ws_size,
                              hipStream_t stream) {
  const float* inp    = (const float*)d_in[0];
  const float* weight = (const float*)d_in[1];
  const float* bias   = (const float*)d_in[2];
  float* out          = (float*)d_out;

  unsigned char* ws   = (unsigned char*)d_ws;
  unsigned* amax      = (unsigned*)ws;
  unsigned char* xq   = ws + 256;
  unsigned char* wqt  = xq + (size_t)NROWS * KDIM;

  fp8lin_init_kernel<<<1, 32, 0, stream>>>(amax);
  fp8lin_amax_kernel<<<2048, 256, 0, stream>>>(inp,    (long)NROWS * KDIM, amax + 0);
  fp8lin_amax_kernel<<<2048, 256, 0, stream>>>(weight, (long)KDIM * NOUT,  amax + 1);
  fp8lin_quant_x_kernel<<<8192, 256, 0, stream>>>(inp, (unsigned*)xq, amax,
                                                  (long)NROWS * KDIM / 4);
  fp8lin_quant_w_kernel<<<dim3(NOUT / 64, KDIM / 64), 256, 0, stream>>>(weight, wqt, amax);
  fp8lin_gemm_kernel<<<dim3(NOUT / BN, NROWS / BM), 256, 0, stream>>>(xq, wqt, bias,
                                                                      amax, out);
}
